// encoder_chain_4956392259718
// MI455X (gfx1250) — compile-verified, matrix operand order checked by reading, not run on hardware
//
#include <hip/hip_runtime.h>

#define NN   50000
#define EE   800000
#define IND  128
#define HID  128
#define OUTD 64
#define NHH  4
#define NEE  2
#define NEG_SLOPE 0.2f
#define NTILES 3125            /* NN/16 exactly */

typedef __attribute__((ext_vector_type(16))) _Float16 v16h;
typedef __attribute__((ext_vector_type(8)))  float    v8f;

// ---------- order-preserving float<->uint encoding for atomicMax ----------
static __device__ __forceinline__ unsigned fenc(float f) {
  int i = __float_as_int(f);
  return ((unsigned)i) ^ (unsigned)((i >> 31) | 0x80000000);
}
static __device__ __forceinline__ float fdec(unsigned u) {
  int i = (u & 0x80000000u) ? (int)(u ^ 0x80000000u) : (int)(~u);
  return __int_as_float(i);
}

// ---------- utility fills ----------
__global__ void k_fill(float* __restrict__ p, float v, long long n) {
  long long i = (long long)blockIdx.x * blockDim.x + threadIdx.x;
  if (i < n) p[i] = v;
}
__global__ void k_fill_neginf(unsigned* __restrict__ p, long long n) {
  long long i = (long long)blockIdx.x * blockDim.x + threadIdx.x;
  if (i < n) p[i] = fenc(-__builtin_inff());
}

// =====================================================================
// WMMA GEMM: C(NxM) = A(NxK) * B(KxM); f16 multiply, f32 accumulate.
// Block = 256 threads (8 waves). The block stages a (K x MT*16) weight
// slice into LDS once, pre-swizzled into the WMMA B-fragment register
// order, so each wave's B fragment is a single 32-byte ds_load. Each wave
// owns one 16-row tile, keeps the A fragment in registers, and sweeps MT
// column tiles with MT register accumulators (MT wmma per K-step).
// =====================================================================
template <int MT>
__global__ void k_gemm_wmma(const float* __restrict__ A, const float* __restrict__ B,
                            float* __restrict__ C, int N, int K, int M) {
  __shared__ _Float16 Wlds[16384];                 // up to 128 x 128 f16 = 32 KB
  const int tid  = threadIdx.x;
  const int wave = tid >> 5;
  const int lane = tid & 31;
  const int cg0  = blockIdx.x * (MT * 16);         // first column of this group

  // ---- stage weight slice into LDS in B-fragment order ----
  // fragment element e for in-block k offset 'kin': e = (kin&7) | ((kin&16)>>1)
  // lane half: (kin>>3)&1  (K 0-7/16-23 -> lanes 0-15 ; K 8-15/24-31 -> lanes 16-31)
  const int nelem = K * MT * 16;
  for (int i = tid; i < nelem; i += 256) {
    const int k   = i / (MT * 16);
    const int c   = i % (MT * 16);
    const int tn  = c >> 4, cl = c & 15;
    const int k0  = k >> 5, kin = k & 31;
    const int hlf = (kin >> 3) & 1;
    const int e   = (kin & 7) | ((kin & 16) >> 1);
    Wlds[((k0 * MT + tn) << 9) + ((cl + (hlf << 4)) << 4) + e] =
        (_Float16)B[(long long)k * M + cg0 + c];
  }
  __syncthreads();

  const int tm = blockIdx.y * 8 + wave;            // row tile (wave-uniform guard)
  if (tm * 16 >= N) return;
  const int row   = tm * 16 + (lane & 15);
  const int kbase = (lane < 16) ? 0 : 8;

  v8f acc[MT];
#pragma unroll
  for (int t = 0; t < MT; ++t) acc[t] = (v8f){};

  for (int k0 = 0; k0 < K; k0 += 32) {
    // A fragment: two contiguous 8-float K segments per lane half
    const float* ap = A + (long long)row * K + k0 + kbase;
    const float4 a0 = *(const float4*)(ap);
    const float4 a1 = *(const float4*)(ap + 4);
    const float4 a2 = *(const float4*)(ap + 16);
    const float4 a3 = *(const float4*)(ap + 20);
    v16h a;
    a[0]  = (_Float16)a0.x; a[1]  = (_Float16)a0.y;
    a[2]  = (_Float16)a0.z; a[3]  = (_Float16)a0.w;
    a[4]  = (_Float16)a1.x; a[5]  = (_Float16)a1.y;
    a[6]  = (_Float16)a1.z; a[7]  = (_Float16)a1.w;
    a[8]  = (_Float16)a2.x; a[9]  = (_Float16)a2.y;
    a[10] = (_Float16)a2.z; a[11] = (_Float16)a2.w;
    a[12] = (_Float16)a3.x; a[13] = (_Float16)a3.y;
    a[14] = (_Float16)a3.z; a[15] = (_Float16)a3.w;

    const int fbase = (k0 >> 5) * MT;
#pragma unroll
    for (int t = 0; t < MT; ++t) {
      const v16h b = *(const v16h*)&Wlds[((fbase + t) << 9) + (lane << 4)];
      acc[t] = __builtin_amdgcn_wmma_f32_16x16x32_f16(false, a, false, b, (short)0,
                                                      acc[t], false, false);
    }
  }

  // C/D layout: VGPR r -> row r (lanes 0-15) / r+8 (lanes 16-31); N = lane%16
  const int rbase = tm * 16 + ((lane >> 4) << 3);
#pragma unroll
  for (int t = 0; t < MT; ++t) {
    const int col = cg0 + t * 16 + (lane & 15);
#pragma unroll
    for (int r = 0; r < 8; ++r)
      C[(long long)(rbase + r) * M + col] = acc[t][r];
  }
}

// ---------- per-(node,head) attention logits ----------
__global__ void k_elr(const float* __restrict__ Z, const float* __restrict__ al,
                      const float* __restrict__ ar, float* __restrict__ el,
                      float* __restrict__ er, int n, int H, int F) {
  int i = blockIdx.x * blockDim.x + threadIdx.x;      // node*H + h
  if (i >= n * H) return;
  const int h = i % H;
  const float* z = Z + (long long)i * F;
  const float* L = al + h * F;
  const float* R = ar + h * F;
  float sl = 0.f, sr = 0.f;
  for (int f = 0; f < F; f += 4) {
    const float4 v = *(const float4*)(z + f);
    const float4 l = *(const float4*)(L + f);
    const float4 r = *(const float4*)(R + f);
    sl += v.x * l.x + v.y * l.y + v.z * l.z + v.w * l.w;
    sr += v.x * r.x + v.y * r.y + v.z * r.z + v.w * r.w;
  }
  el[i] = sl; er[i] = sr;
}

// ---------- edge softmax pass 1: segment max ----------
__global__ void k_edge_max(const int* __restrict__ src, const int* __restrict__ dst,
                           const float* __restrict__ el, const float* __restrict__ er,
                           unsigned* __restrict__ m, long long EH, int H) {
  long long i = (long long)blockIdx.x * blockDim.x + threadIdx.x;
  if (i >= EH) return;
  const long long e = i / H; const int h = (int)(i % H);
  const int s = src[e], d = dst[e];
  float x = el[s * H + h] + er[d * H + h];
  x = (x >= 0.f) ? x : NEG_SLOPE * x;
  atomicMax(&m[d * H + h], fenc(x));
}

// ---------- edge softmax pass 2: exp + segment denominator ----------
__global__ void k_edge_exp(const int* __restrict__ src, const int* __restrict__ dst,
                           const float* __restrict__ el, const float* __restrict__ er,
                           const unsigned* __restrict__ m, float* __restrict__ exb,
                           float* __restrict__ den, long long EH, int H) {
  long long i = (long long)blockIdx.x * blockDim.x + threadIdx.x;
  if (i >= EH) return;
  const long long e = i / H; const int h = (int)(i % H);
  const int s = src[e], d = dst[e];
  float x = el[s * H + h] + er[d * H + h];
  x = (x >= 0.f) ? x : NEG_SLOPE * x;
  float mx = fdec(m[d * H + h]);
  if (!(mx > -3.0e38f && mx < 3.0e38f)) mx = 0.f;     // reference's finite-m safety
  const float ex = __expf(x - mx);
  exb[i] = ex;
  atomicAdd(&den[d * H + h], ex);
}

// ---------- weighted scatter: num[dst,h,f..f+3] += ex * z[src,h,f..f+3] ----------
__global__ void k_edge_scatter(const int* __restrict__ src, const int* __restrict__ dst,
                               const float* __restrict__ Z, const float* __restrict__ exb,
                               float* __restrict__ num, long long EF4, int H, int F) {
  long long i = (long long)blockIdx.x * blockDim.x + threadIdx.x;
  if (i >= EF4) return;
  const int fq = F >> 2;
  const long long e = i / fq;
  const int f = (int)(i % fq) << 2;
  const long long ne = EF4 / fq;
  const int s = src[e], d = dst[e];
  if (e + 4 < ne)                                     // warm L2 for upcoming gathers
    __builtin_prefetch(&Z[(long long)src[e + 4] * H * F], 0, 1);
  for (int h = 0; h < H; ++h) {
    const float ex = exb[e * H + h];
    const float4 z = *(const float4*)&Z[((long long)s * H + h) * F + f];
    float* o = &num[((long long)d * H + h) * F + f];
    atomicAdd(o + 0, ex * z.x);
    atomicAdd(o + 1, ex * z.y);
    atomicAdd(o + 2, ex * z.z);
    atomicAdd(o + 3, ex * z.w);
  }
}

// ---------- normalize + bias, scale, accumulate into layer output ----------
__global__ void k_combine(float* __restrict__ acc, const float* __restrict__ num,
                          const float* __restrict__ den, const float* __restrict__ bias,
                          const float* __restrict__ fcw, int etype, int use_fc,
                          long long NHF, int H, int F) {
  long long i = (long long)blockIdx.x * blockDim.x + threadIdx.x;
  if (i >= NHF) return;
  const int hf = (int)(i % (H * F));
  const long long nh = i / F;
  const float v = num[i] / fmaxf(den[nh], 1e-9f) + bias[hf];
  // mean over etypes -> 0.5 ; layer2: mean_f of fc_w[f][e] applied per etype
  const float s = use_fc ? 0.5f * (fcw[etype] + fcw[NEE + etype]) : 0.5f;
  acc[i] += s * v;
}

// ---------- column sums for mean-centering (LDS tree reduction) ----------
__global__ void k_colsum(const float* __restrict__ hm, float* __restrict__ colsum,
                         int n, int HF) {
  const int hf = blockIdx.x;
  float s = 0.f;
  for (int nd = threadIdx.x; nd < n; nd += blockDim.x) s += hm[(long long)nd * HF + hf];
  __shared__ float red[256];
  red[threadIdx.x] = s;
  __syncthreads();
  for (int w = 128; w > 0; w >>= 1) {
    if (threadIdx.x < w) red[threadIdx.x] += red[threadIdx.x + w];
    __syncthreads();
  }
  if (threadIdx.x == 0) colsum[hf] = red[0];
}

__global__ void k_center(const float* __restrict__ hm, const float* __restrict__ colsum,
                         float* __restrict__ out, long long n, int HF, float invN) {
  long long i = (long long)blockIdx.x * blockDim.x + threadIdx.x;
  if (i < n) out[i] = hm[i] - colsum[i % HF] * invN;
}

// =====================================================================
static inline unsigned blks(long long n, int b) { return (unsigned)((n + b - 1) / b); }

static void launch_gemm(const float* A, const float* B, float* C, int K, int M,
                        hipStream_t st) {
  const unsigned gy = (NTILES + 7) / 8;
  if ((M & 127) == 0) {        // M = 128 or 256: MT=8, M/128 column groups
    dim3 g((unsigned)(M / 128), gy);
    k_gemm_wmma<8><<<g, 256, 0, st>>>(A, B, C, NN, K, M);
  } else {                     // M = 64: MT=4, one group
    dim3 g(1, gy);
    k_gemm_wmma<4><<<g, 256, 0, st>>>(A, B, C, NN, K, M);
  }
}

static void run_gat_etype(const float* Hin, int K, const float* W, const float* al,
                          const float* ar, const float* bias, int H, int F,
                          const int* src, const int* dst,
                          float* Z, float* num, float* el, float* er, unsigned* m,
                          float* den, float* exb, float* acc, const float* fcw,
                          int use_fc, int etype, hipStream_t st) {
  const int M = H * F;
  const long long nM = (long long)NN * M;
  const long long nH = (long long)NN * H;
  k_fill<<<blks(nM, 256), 256, 0, st>>>(num, 0.f, nM);
  k_fill<<<blks(nH, 256), 256, 0, st>>>(den, 0.f, nH);
  k_fill_neginf<<<blks(nH, 256), 256, 0, st>>>(m, nH);

  launch_gemm(Hin, W, Z, K, M, st);

  k_elr<<<blks(nH, 256), 256, 0, st>>>(Z, al, ar, el, er, NN, H, F);

  const long long EH = (long long)EE * H;
  k_edge_max<<<blks(EH, 256), 256, 0, st>>>(src, dst, el, er, m, EH, H);
  k_edge_exp<<<blks(EH, 256), 256, 0, st>>>(src, dst, el, er, m, exb, den, EH, H);

  const long long EF4 = (long long)EE * (F >> 2);
  k_edge_scatter<<<blks(EF4, 256), 256, 0, st>>>(src, dst, Z, exb, num, EF4, H, F);

  k_combine<<<blks(nM, 256), 256, 0, st>>>(acc, num, den, bias, fcw, etype, use_fc,
                                           nM, H, F);
}

extern "C" void kernel_launch(void* const* d_in, const int* in_sizes, int n_in,
                              void* d_out, int out_size, void* d_ws, size_t ws_size,
                              hipStream_t stream) {
  (void)in_sizes; (void)n_in; (void)out_size; (void)ws_size;
  const float* x      = (const float*)d_in[0];
  const int*   src[2] = {(const int*)d_in[1], (const int*)d_in[3]};
  const int*   dst[2] = {(const int*)d_in[2], (const int*)d_in[4]};
  const float* W1  = (const float*)d_in[5];
  const float* al1 = (const float*)d_in[6];
  const float* ar1 = (const float*)d_in[7];
  const float* b1  = (const float*)d_in[8];
  const float* W2  = (const float*)d_in[9];
  const float* al2 = (const float*)d_in[10];
  const float* ar2 = (const float*)d_in[11];
  const float* b2  = (const float*)d_in[12];
  const float* Wm  = (const float*)d_in[13];
  const float* alm = (const float*)d_in[14];
  const float* arm = (const float*)d_in[15];
  const float* bm  = (const float*)d_in[16];
  const float* fcw = (const float*)d_in[17];

  // workspace layout (floats)
  float* ws = (float*)d_ws;
  size_t off = 0;
  float*    Z    = ws + off; off += (size_t)NN * 256;      // projected features (max M)
  float*    num  = ws + off; off += (size_t)NN * 256;      // scatter accumulators
  float*    el   = ws + off; off += (size_t)NN * NHH;
  float*    er   = ws + off; off += (size_t)NN * NHH;
  unsigned* mEnc = (unsigned*)(ws + off); off += (size_t)NN * NHH;
  float*    den  = ws + off; off += (size_t)NN * NHH;
  float*    exb  = ws + off; off += (size_t)EE * NHH;
  float*    h1   = ws + off; off += (size_t)NN * HID;
  float*    h2   = ws + off; off += (size_t)NN * OUTD;
  float*    hm   = ws + off; off += (size_t)NN * NHH * OUTD;
  float*    csum = ws + off; off += 256;

  // ---- layer 1: mean over etypes, H=1, F=HID ----
  k_fill<<<blks((long long)NN * HID, 256), 256, 0, stream>>>(h1, 0.f, (long long)NN * HID);
  for (int e = 0; e < NEE; ++e)
    run_gat_etype(x, IND, W1 + (size_t)e * IND * HID, al1 + (size_t)e * HID,
                  ar1 + (size_t)e * HID, b1 + (size_t)e * HID, 1, HID,
                  src[e], dst[e], Z, num, el, er, mEnc, den, exb, h1,
                  fcw, 0, e, stream);

  // ---- layer 2: fc over etype axis then mean, H=1, F=OUT ----
  k_fill<<<blks((long long)NN * OUTD, 256), 256, 0, stream>>>(h2, 0.f, (long long)NN * OUTD);
  for (int e = 0; e < NEE; ++e)
    run_gat_etype(h1, HID, W2 + (size_t)e * HID * OUTD, al2 + (size_t)e * OUTD,
                  ar2 + (size_t)e * OUTD, b2 + (size_t)e * OUTD, 1, OUTD,
                  src[e], dst[e], Z, num, el, er, mEnc, den, exb, h2,
                  fcw, 1, e, stream);

  // ---- multi-head layer: mean over etypes, H=NH, F=OUT ----
  const long long nhm = (long long)NN * NHH * OUTD;
  k_fill<<<blks(nhm, 256), 256, 0, stream>>>(hm, 0.f, nhm);
  for (int e = 0; e < NEE; ++e)
    run_gat_etype(h2, OUTD, Wm + (size_t)e * OUTD * NHH * OUTD,
                  alm + (size_t)e * NHH * OUTD, arm + (size_t)e * NHH * OUTD,
                  bm + (size_t)e * NHH * OUTD, NHH, OUTD,
                  src[e], dst[e], Z, num, el, er, mEnc, den, exb, hm,
                  fcw, 0, e, stream);

  // ---- per-head per-feature mean-centering over nodes ----
  k_colsum<<<NHH * OUTD, 256, 0, stream>>>(hm, csum, NN, NHH * OUTD);
  k_center<<<blks(nhm, 256), 256, 0, stream>>>(hm, csum, (float*)d_out, nhm,
                                               NHH * OUTD, 1.0f / (float)NN);
}
